// ALiBiAttention_73065983639812
// MI455X (gfx1250) — matrix operand order repaired and checked by reading, not compiled, not run on hardware
//
#include <hip/hip_runtime.h>
#include <hip/hip_bf16.h>

typedef __attribute__((ext_vector_type(16))) __bf16 v16bf;
typedef __attribute__((ext_vector_type(8)))  float  v8f;
typedef __attribute__((ext_vector_type(4)))  int    v4i;

union Frag { v16bf v; uint4 u[2]; };

static __device__ __forceinline__ v8f wmma_bf16f32(v16bf a, v16bf b, v8f c) {
  // D(f32 16x16) = A(bf16 16x32) * B(bf16 32x16) + C
  return __builtin_amdgcn_wmma_f32_16x16x32_bf16(false, a, false, b, (short)0, c,
                                                 false, false);
}

// -------- gfx1250 async Global->LDS staging (ASYNCcnt path) ----------------
#if __has_builtin(__builtin_amdgcn_global_load_async_to_lds_b128) && \
    __has_builtin(__builtin_amdgcn_s_wait_asynccnt)
#define USE_ASYNC_LDS 1
#else
#define USE_ASYNC_LDS 0
#endif

typedef __attribute__((address_space(1))) v4i gv4i;  // global int4-vector
typedef __attribute__((address_space(3))) v4i lv4i;  // LDS int4-vector

static __device__ __forceinline__ void cp16_to_lds(void* l, const void* g) {
#if USE_ASYNC_LDS
  __builtin_amdgcn_global_load_async_to_lds_b128((gv4i*)g, (lv4i*)l, 0, 0);
#else
  *(uint4*)l = *(const uint4*)g;
#endif
}
static __device__ __forceinline__ void cp_lds_wait() {
#if USE_ASYNC_LDS
  __builtin_amdgcn_s_wait_asynccnt(0);
#endif
}

constexpr int BATCH  = 4;
constexpr int SEQ    = 2048;
constexpr int DMODEL = 1024;
constexpr int NH     = 16;
constexpr int HD     = 64;
constexpr int MTOT   = BATCH * SEQ;   // 8192
constexpr int KTOT   = DMODEL;
constexpr int NTOT   = DMODEL;

// ---------------------------------------------------------------------------
// fp32 -> bf16 conversion
// ---------------------------------------------------------------------------
__global__ __launch_bounds__(256)
void cvt_f32_bf16(const float* __restrict__ s, __bf16* __restrict__ d, int n) {
  int i  = blockIdx.x * blockDim.x + threadIdx.x;
  int st = gridDim.x * blockDim.x;
  for (; i < n; i += st) d[i] = (__bf16)s[i];
}

// ---------------------------------------------------------------------------
// bf16 WMMA GEMM:  Y[m,n] = (sum_k A[m,k] * W[n,k] + bias[n]) * scl
// OUT_MODE 0: store bf16 head-major [B,H,S,hd]   (QKV projections)
// OUT_MODE 1: store f32 row-major [M,N]          (output projection)
// Block: 256 thr = 8 waves (4 in M x 2 in N); tile 128x128, K-step 32.
// ---------------------------------------------------------------------------
template <int OUT_MODE>
__global__ __launch_bounds__(256)
void gemm_bf16_wmma(const __bf16* __restrict__ A, const __bf16* __restrict__ Bw,
                    const float* __restrict__ bias, float scl,
                    void* __restrict__ out) {
  constexpr int BM = 128, BN = 128, KT = 32, LDT = KT + 8;  // 80B row stride
  __shared__ __bf16 As[BM][LDT];
  __shared__ __bf16 Bs[BN][LDT];

  const int t     = threadIdx.x;
  const int lane  = t & 31, wid = t >> 5;
  const int waveM = wid & 3, waveN = wid >> 2;
  const int half  = lane >> 4, col_l = lane & 15;
  const int m0 = blockIdx.y * BM, n0 = blockIdx.x * BN;

  v8f acc[2][4];
#pragma unroll
  for (int mf = 0; mf < 2; mf++)
#pragma unroll
    for (int nf = 0; nf < 4; nf++)
#pragma unroll
      for (int j = 0; j < 8; j++) acc[mf][nf][j] = 0.f;

  for (int kb = 0; kb < KTOT; kb += KT) {
    __syncthreads();
    // Stage 128x32 bf16 tiles of A and W into LDS (async DMA when available).
#pragma unroll
    for (int i = 0; i < 2; i++) {
      int v = t + i * 256;
      int r = v >> 2, c = (v & 3) * 8;
      cp16_to_lds(&As[r][c], A  + (size_t)(m0 + r) * KTOT + kb + c);
      cp16_to_lds(&Bs[r][c], Bw + (size_t)(n0 + r) * KTOT + kb + c);
      if (kb + KT < KTOT) {
        __builtin_prefetch(A  + (size_t)(m0 + r) * KTOT + kb + KT + c, 0, 0);
        __builtin_prefetch(Bw + (size_t)(n0 + r) * KTOT + kb + KT + c, 0, 0);
      }
    }
    cp_lds_wait();
    __syncthreads();

    // Fragment loads per the gfx1250 16-bit A(16x32)/B(32x16) lane layouts.
    Frag af[2], bf[4];
#pragma unroll
    for (int mf = 0; mf < 2; mf++) {
      int r = waveM * 32 + mf * 16 + col_l;          // row M = lane&15
      af[mf].u[0] = *(const uint4*)&As[r][half * 8];       // K = h*8 .. h*8+7
      af[mf].u[1] = *(const uint4*)&As[r][half * 8 + 16];  // K = h*8+16 .. +23
    }
#pragma unroll
    for (int nf = 0; nf < 4; nf++) {
      int r = waveN * 64 + nf * 16 + col_l;          // col N = lane&15 (row of W)
      bf[nf].u[0] = *(const uint4*)&Bs[r][half * 8];       // K = h*8 .. h*8+7
      bf[nf].u[1] = *(const uint4*)&Bs[r][half * 8 + 16];  // K = h*8+16 .. +23
    }
#pragma unroll
    for (int mf = 0; mf < 2; mf++)
#pragma unroll
      for (int nf = 0; nf < 4; nf++)
        acc[mf][nf] = wmma_bf16f32(af[mf].v, bf[nf].v, acc[mf][nf]);
  }

  float b4[4];
#pragma unroll
  for (int nf = 0; nf < 4; nf++) b4[nf] = bias[n0 + waveN * 64 + nf * 16 + col_l];

#pragma unroll
  for (int mf = 0; mf < 2; mf++)
#pragma unroll
    for (int nf = 0; nf < 4; nf++) {
      const int colg = n0 + waveN * 64 + nf * 16 + col_l;
#pragma unroll
      for (int j = 0; j < 8; j++) {
        // C/D layout: VGPR j of lane holds (M = 8*(lane>>4)+j, N = lane&15)
        const int rowg = m0 + waveM * 32 + mf * 16 + half * 8 + j;
        const float vv = (acc[mf][nf][j] + b4[nf]) * scl;
        if (OUT_MODE == 0) {
          const int bb = rowg >> 11, ss = rowg & (SEQ - 1);
          const int hh = colg >> 6,  dd = colg & (HD - 1);
          ((__bf16*)out)[(((size_t)bb * NH + hh) * SEQ + ss) * HD + dd] = (__bf16)vv;
        } else {
          ((float*)out)[(size_t)rowg * NTOT + colg] = vv;
        }
      }
    }
}

// ---------------------------------------------------------------------------
// Fused ALiBi flash attention (non-causal, masked, online softmax).
// Block: 256 thr = 8 waves, each wave owns 16 q-rows; q-tile 128, k-tile 64.
// Q (pre-scaled by hd^-0.5), K, V in [B,H,S,hd] bf16; ctx out [B,S,D] bf16.
// ---------------------------------------------------------------------------
__global__ __launch_bounds__(256)
void attn_alibi_wmma(const __bf16* __restrict__ Qh, const __bf16* __restrict__ Kh,
                     const __bf16* __restrict__ Vh, const int* __restrict__ mask,
                     __bf16* __restrict__ ctx) {
  constexpr int BQ = 128, BK = 64, LDK = HD + 16;  // 160B rows: 16B-aligned b128
  __shared__ __bf16 Ks[BK][LDK];     // K tile, [s][d] (rows contiguous in d)
  __shared__ __bf16 VTs[HD][LDK];    // V tile transposed, [d][s]
  __shared__ float  mb[BK];          // mask bias per key
  __shared__ __bf16 Ps[8][16][LDK];  // per-wave P re-layout buffer

  const int bh = blockIdx.x;
  const int b = bh >> 4, h = bh & (NH - 1);
  const int q0 = blockIdx.y * BQ;
  const int t = threadIdx.x, lane = t & 31, wid = t >> 5;
  const int half = lane >> 4, col_l = lane & 15;
  const size_t hoff = (size_t)bh * SEQ * HD;
  const __bf16* Qp = Qh + hoff;
  const __bf16* Kp = Kh + hoff;
  const __bf16* Vp = Vh + hoff;

  // Q A-fragments for both K-steps over hd=64 (resident for the whole pass).
  Frag aq[2];
  {
    const int qr = q0 + wid * 16 + col_l;
#pragma unroll
    for (int ks = 0; ks < 2; ks++) {
      aq[ks].u[0] = *(const uint4*)(Qp + (size_t)qr * HD + ks * 32 + half * 8);
      aq[ks].u[1] = *(const uint4*)(Qp + (size_t)qr * HD + ks * 32 + half * 8 + 16);
    }
  }

  const float slope = exp2f(-(float)h / (float)NH);  // 1 / 2^(h/H)

  // ALiBi is affine: hoist slope*q (per row) and slope*col (per lane).
  float rowterm[8];
#pragma unroll
  for (int j = 0; j < 8; j++)
    rowterm[j] = slope * (float)(q0 + wid * 16 + half * 8 + j);
  float colbase[4];
#pragma unroll
  for (int nf = 0; nf < 4; nf++)
    colbase[nf] = slope * (float)(nf * 16 + col_l);

  float m_r[8], l_r[8];
  v8f o[4];
#pragma unroll
  for (int j = 0; j < 8; j++) { m_r[j] = -1e30f; l_r[j] = 0.f; }
#pragma unroll
  for (int nf = 0; nf < 4; nf++)
#pragma unroll
    for (int j = 0; j < 8; j++) o[nf][j] = 0.f;

  for (int kt = 0; kt < SEQ; kt += BK) {
    __syncthreads();
    // Stage K tile (async DMA), and V tile transposed through VGPRs.
#pragma unroll
    for (int i = 0; i < 2; i++) {
      int v = t + i * 256;
      int r = v >> 3, c = (v & 7) * 8;
      cp16_to_lds(&Ks[r][c], Kp + (size_t)(kt + r) * HD + c);
      union { uint4 q; __bf16 e[8]; } tv;
      tv.q = *(const uint4*)(Vp + (size_t)(kt + r) * HD + c);
#pragma unroll
      for (int ee = 0; ee < 8; ee++) VTs[c + ee][r] = tv.e[ee];
    }
    if (t < BK) mb[t] = mask[b * SEQ + kt + t] ? 0.f : -1e30f;
    cp_lds_wait();
    __syncthreads();

    // S = Q * K^T : B-frag column n = key index -> K row (contiguous in hd).
    v8f sc[4];
#pragma unroll
    for (int nf = 0; nf < 4; nf++) {
#pragma unroll
      for (int j = 0; j < 8; j++) sc[nf][j] = 0.f;
      const int r = nf * 16 + col_l;
#pragma unroll
      for (int ks = 0; ks < 2; ks++) {
        Frag kb;
        kb.u[0] = *(const uint4*)&Ks[r][ks * 32 + half * 8];
        kb.u[1] = *(const uint4*)&Ks[r][ks * 32 + half * 8 + 16];
        sc[nf] = wmma_bf16f32(aq[ks].v, kb.v, sc[nf]);
      }
    }

    // + ALiBi + mask (Q already carries the hd^-0.5 scale).
    float cterm[4];
#pragma unroll
    for (int nf = 0; nf < 4; nf++)
      cterm[nf] = colbase[nf] + slope * (float)kt + mb[nf * 16 + col_l];
#pragma unroll
    for (int nf = 0; nf < 4; nf++)
#pragma unroll
      for (int j = 0; j < 8; j++)
        sc[nf][j] = sc[nf][j] + (cterm[nf] - rowterm[j]);

    // Online softmax: rows live in 16-lane halves -> shfl_xor reductions.
#pragma unroll
    for (int j = 0; j < 8; j++) {
      float mx = fmaxf(fmaxf(sc[0][j], sc[1][j]), fmaxf(sc[2][j], sc[3][j]));
#pragma unroll
      for (int d = 1; d < 16; d <<= 1) mx = fmaxf(mx, __shfl_xor(mx, d, 32));
      const float mnew = fmaxf(m_r[j], mx);
      const float corr = __expf(m_r[j] - mnew);
      float rs = 0.f;
#pragma unroll
      for (int nf = 0; nf < 4; nf++) {
        const float p = __expf(sc[nf][j] - mnew);
        sc[nf][j] = p;
        rs += p;
      }
#pragma unroll
      for (int d = 1; d < 16; d <<= 1) rs += __shfl_xor(rs, d, 32);
      l_r[j] = l_r[j] * corr + rs;
      m_r[j] = mnew;
#pragma unroll
      for (int nf = 0; nf < 4; nf++) o[nf][j] *= corr;
    }

    // Re-layout P (C layout -> A layout) through per-wave LDS scratch.
#pragma unroll
    for (int nf = 0; nf < 4; nf++)
#pragma unroll
      for (int j = 0; j < 8; j++)
        Ps[wid][half * 8 + j][nf * 16 + col_l] = (__bf16)sc[nf][j];

    // O += P * V  (B-frag from V^T: contiguous along key dim).
#pragma unroll
    for (int ks2 = 0; ks2 < 2; ks2++) {
      Frag ap;
      ap.u[0] = *(const uint4*)&Ps[wid][col_l][ks2 * 32 + half * 8];
      ap.u[1] = *(const uint4*)&Ps[wid][col_l][ks2 * 32 + half * 8 + 16];
#pragma unroll
      for (int nf = 0; nf < 4; nf++) {
        Frag bv;
        const int vr = nf * 16 + col_l;
        bv.u[0] = *(const uint4*)&VTs[vr][ks2 * 32 + half * 8];
        bv.u[1] = *(const uint4*)&VTs[vr][ks2 * 32 + half * 8 + 16];
        o[nf] = wmma_bf16f32(ap.v, bv.v, o[nf]);
      }
    }
  }

  // Normalize and write ctx as bf16 [B,S,D] (row-major input to final GEMM).
#pragma unroll
  for (int nf = 0; nf < 4; nf++) {
    const int c = h * HD + nf * 16 + col_l;
#pragma unroll
    for (int j = 0; j < 8; j++) {
      const int r = q0 + wid * 16 + half * 8 + j;
      const float vv = o[nf][j] / l_r[j];
      ctx[((size_t)b * SEQ + r) * DMODEL + c] = (__bf16)vv;
    }
  }
}

// ---------------------------------------------------------------------------
extern "C" void kernel_launch(void* const* d_in, const int* in_sizes, int n_in,
                              void* d_out, int out_size, void* d_ws, size_t ws_size,
                              hipStream_t stream) {
  (void)in_sizes; (void)n_in; (void)out_size; (void)ws_size;

  const float* x   = (const float*)d_in[0];
  const float* Wq  = (const float*)d_in[1];
  const float* bq  = (const float*)d_in[2];
  const float* Wk  = (const float*)d_in[3];
  const float* bk  = (const float*)d_in[4];
  const float* Wv  = (const float*)d_in[5];
  const float* bv  = (const float*)d_in[6];
  const float* Wo  = (const float*)d_in[7];
  const float* bo  = (const float*)d_in[8];
  const int*  mask = (const int*)d_in[9];

  const size_t XE = (size_t)MTOT * DMODEL;    // 8Mi elems
  const size_t WE = (size_t)DMODEL * DMODEL;  // 1Mi elems
  char* p = (char*)d_ws;
  __bf16* xb  = (__bf16*)p; p += XE * 2;
  __bf16* wqb = (__bf16*)p; p += WE * 2;
  __bf16* wkb = (__bf16*)p; p += WE * 2;
  __bf16* wvb = (__bf16*)p; p += WE * 2;
  __bf16* wob = (__bf16*)p; p += WE * 2;
  __bf16* Qh  = (__bf16*)p; p += XE * 2;
  __bf16* Kh  = (__bf16*)p; p += XE * 2;
  __bf16* Vh  = (__bf16*)p; p += XE * 2;
  __bf16* ctx = (__bf16*)p; p += XE * 2;     // total ~88 MB of workspace

  cvt_f32_bf16<<<4096, 256, 0, stream>>>(x,  xb,  (int)XE);
  cvt_f32_bf16<<<1024, 256, 0, stream>>>(Wq, wqb, (int)WE);
  cvt_f32_bf16<<<1024, 256, 0, stream>>>(Wk, wkb, (int)WE);
  cvt_f32_bf16<<<1024, 256, 0, stream>>>(Wv, wvb, (int)WE);
  cvt_f32_bf16<<<1024, 256, 0, stream>>>(Wo, wob, (int)WE);

  const dim3 gg(NTOT / 128, MTOT / 128);  // (8, 64)
  const float qscale = 0.125f;            // hd^-0.5 folded into Q projection
  gemm_bf16_wmma<0><<<gg, 256, 0, stream>>>(xb, wqb, bq, qscale, (void*)Qh);
  gemm_bf16_wmma<0><<<gg, 256, 0, stream>>>(xb, wkb, bk, 1.0f,   (void*)Kh);
  gemm_bf16_wmma<0><<<gg, 256, 0, stream>>>(xb, wvb, bv, 1.0f,   (void*)Vh);

  attn_alibi_wmma<<<dim3(BATCH * NH, SEQ / 128), 256, 0, stream>>>(Qh, Kh, Vh, mask, ctx);

  gemm_bf16_wmma<1><<<gg, 256, 0, stream>>>(ctx, wob, bo, 1.0f, d_out);
}
